// Attention_60292750901466
// MI455X (gfx1250) — matrix operand order repaired and checked by reading, not compiled, not run on hardware
//
#include <hip/hip_runtime.h>
#include <hip/hip_bf16.h>

// ---------------------------------------------------------------------------
// Shapes (fixed by the reference)
// ---------------------------------------------------------------------------
#define Bz   4
#define Sq   1024
#define Dm   4096
#define NH   32
#define NKV  8
#define HDm  128
#define REP  4          // NH / NKV

typedef __bf16 bf16_t;
typedef __attribute__((ext_vector_type(8)))  __bf16 v8bf;
typedef __attribute__((ext_vector_type(16))) __bf16 v16bf;
typedef __attribute__((ext_vector_type(8)))  float  v8f;
typedef __attribute__((ext_vector_type(4)))  int    v4i;

#define CAT16(lo, hi) __builtin_shufflevector((lo), (hi), 0,1,2,3,4,5,6,7,8,9,10,11,12,13,14,15)

// ---------------------------------------------------------------------------
// CDNA5 async global->LDS copy (ASYNCcnt) with sync fallback
// ---------------------------------------------------------------------------
#if defined(__has_builtin)
#if __has_builtin(__builtin_amdgcn_global_load_async_to_lds_b128) && \
    __has_builtin(__builtin_amdgcn_s_wait_asynccnt)
#define HAVE_ASYNC_LDS 1
#endif
#endif
#ifndef HAVE_ASYNC_LDS
#define HAVE_ASYNC_LDS 0
#endif

typedef __attribute__((address_space(1))) v4i* as1_v4i;
typedef __attribute__((address_space(3))) v4i* as3_v4i;

__device__ __forceinline__ void async_cp16(const bf16_t* g, bf16_t* l) {
#if HAVE_ASYNC_LDS
    __builtin_amdgcn_global_load_async_to_lds_b128((as1_v4i)g, (as3_v4i)l, 0, 0);
#else
    *(v8bf*)l = *(const v8bf*)g;
#endif
}
__device__ __forceinline__ void async_wait_all() {
#if HAVE_ASYNC_LDS
    __builtin_amdgcn_s_wait_asynccnt(0);
#endif
}

__device__ __forceinline__ bf16_t f2bf(float f) {
    unsigned u = __builtin_bit_cast(unsigned, f);
    unsigned r = u + 0x7FFFu + ((u >> 16) & 1u);   // round-to-nearest-even
    unsigned short h = (unsigned short)(r >> 16);
    return __builtin_bit_cast(bf16_t, h);
}

__device__ __forceinline__ v8f wmma_bf16(v16bf a, v16bf b, v8f c) {
    return __builtin_amdgcn_wmma_f32_16x16x32_bf16(false, a, false, b, (short)0, c, false, false);
}

// A-fragment (16x32 bf16, MxK): lane holds row = lane%16; K chunks at
// {half*8 .. +8} and {16+half*8 .. +8}  (half = lane/16)
__device__ __forceinline__ v16bf ldA(const bf16_t* rowp, int half8) {
    v8bf lo = *(const v8bf*)(rowp + half8);
    v8bf hi = *(const v8bf*)(rowp + 16 + half8);
    return CAT16(lo, hi);
}
// B-fragment (32x16 bf16, KxN): lane holds col = lane%16; K chunks at
// {half*8 .. +8} and {16+half*8 .. +8}  (half = lane/16)
__device__ __forceinline__ v16bf ldB(const bf16_t* colp, int half16) {
    v8bf lo = *(const v8bf*)(colp + half16 / 2);
    v8bf hi = *(const v8bf*)(colp + 16 + half16 / 2);
    return CAT16(lo, hi);
}

// ---------------------------------------------------------------------------
// Pre-conversion kernels (run once; GEMM hot loops then touch only bf16)
// ---------------------------------------------------------------------------
__global__ __launch_bounds__(256) void cvt_bf16(
    const float* __restrict__ in, bf16_t* __restrict__ out)
{
    size_t i = ((size_t)blockIdx.x * 256 + threadIdx.x) * 8;
    float4 a = *(const float4*)(in + i);
    float4 b = *(const float4*)(in + i + 4);
    v8bf o;
    o[0] = f2bf(a.x); o[1] = f2bf(a.y); o[2] = f2bf(a.z); o[3] = f2bf(a.w);
    o[4] = f2bf(b.x); o[5] = f2bf(b.y); o[6] = f2bf(b.z); o[7] = f2bf(b.w);
    *(v8bf*)(out + i) = o;
}

// W fp32 [K][N] -> Wt bf16 [N][K]   (64x64 tiles through LDS)
__global__ __launch_bounds__(256) void cvt_transpose(
    const float* __restrict__ in, bf16_t* __restrict__ out, int K, int N)
{
    __shared__ float tile[64][65];
    const int kb = blockIdx.x * 64, nb = blockIdx.y * 64;
    const int t = threadIdx.x;
#pragma unroll
    for (int e = 0; e < 16; ++e) {
        int idx = e * 256 + t;
        int r = idx >> 6, c = idx & 63;
        tile[r][c] = in[(size_t)(kb + r) * N + nb + c];
    }
    __syncthreads();
#pragma unroll
    for (int e = 0; e < 16; ++e) {
        int idx = e * 256 + t;
        int nn = idx >> 6, kk = idx & 63;
        out[(size_t)(nb + nn) * K + kb + kk] = f2bf(tile[kk][nn]);
    }
}

// ---------------------------------------------------------------------------
// Tile staging for GEMM: 128x64 bf16 tile (A: [row][K], Bt: [col][K])
// ---------------------------------------------------------------------------
__device__ __forceinline__ void gemm_issue(const bf16_t* __restrict__ Ab,
                                           const bf16_t* __restrict__ Bb,
                                           unsigned char* smem, int cur,
                                           int t, int k0)
{
    bf16_t* As = (bf16_t*)(smem + cur * 32768);
    bf16_t* Bs = (bf16_t*)(smem + cur * 32768 + 16384);
#pragma unroll
    for (int e = 0; e < 4; ++e) {
        int idx = e * 256 + t;          // 1024 chunks of 8 bf16
        int row = idx >> 3, ko = (idx & 7) * 8;
        async_cp16(Ab + (size_t)row * Dm + k0 + ko, &As[row * 64 + ko]);
        async_cp16(Bb + (size_t)row * Dm + k0 + ko, &Bs[row * 64 + ko]);
    }
}

// ---------------------------------------------------------------------------
// Kernel 1: C = seq_b(bf16) @ Wt(bf16,[N][K]) with fused RoPE epilogue;
// output bf16 laid out [b][head][s][hd]. Block tile 128x128, BK=64,
// double-buffered async LDS staging, 8 waves (4x2), wave tile 32x64.
// ---------------------------------------------------------------------------
__global__ __launch_bounds__(256) void qkv_gemm_rope(
    const bf16_t* __restrict__ A, const bf16_t* __restrict__ Wt,
    const float* __restrict__ cosp, const float* __restrict__ sinp,
    bf16_t* __restrict__ out, int nheads, int applyRope)
{
    __shared__ __align__(16) unsigned char smem[128 * 136 * 4]; // >= 64KB tiles
    float* Cst = (float*)smem;            // [128][136] epilogue staging (alias)

    const int t = threadIdx.x;
    const int w = t >> 5, l = t & 31;
    const int wm = w & 3, wn = w >> 2;
    const int lane16 = l & 15;
    const int half8  = ((l >> 4) & 1) * 8;
    const int half16 = ((l >> 4) & 1) * 16;
    const int rowBase = blockIdx.x * 128;
    const int colBase = blockIdx.y * 128;
    const bf16_t* Ab = A  + (size_t)rowBase * Dm;
    const bf16_t* Bb = Wt + (size_t)colBase * Dm;

    v8f acc[2][4];
    const v8f vzero = {0.f,0.f,0.f,0.f,0.f,0.f,0.f,0.f};
#pragma unroll
    for (int mt = 0; mt < 2; ++mt)
#pragma unroll
        for (int nt = 0; nt < 4; ++nt) acc[mt][nt] = vzero;

    gemm_issue(Ab, Bb, smem, 0, t, 0);
    const int NIT = Dm / 64;
    for (int it = 0; it < NIT; ++it) {
        int cur = it & 1;
        async_wait_all();
        __syncthreads();                       // staged tile visible; prev reads done
        if (it + 1 < NIT) gemm_issue(Ab, Bb, smem, cur ^ 1, t, (it + 1) * 64);

        const bf16_t* Asb = (const bf16_t*)(smem + cur * 32768);
        const bf16_t* Bsb = (const bf16_t*)(smem + cur * 32768 + 16384);
#pragma unroll
        for (int kk = 0; kk < 2; ++kk) {
            v16bf af[2], bfr[4];
#pragma unroll
            for (int mt = 0; mt < 2; ++mt)
                af[mt] = ldA(&Asb[(wm * 32 + mt * 16 + lane16) * 64 + kk * 32], half8);
#pragma unroll
            for (int nt = 0; nt < 4; ++nt)
                bfr[nt] = ldB(&Bsb[(wn * 64 + nt * 16 + lane16) * 64 + kk * 32], half16);
#pragma unroll
            for (int mt = 0; mt < 2; ++mt)
#pragma unroll
                for (int nt = 0; nt < 4; ++nt)
                    acc[mt][nt] = wmma_bf16(af[mt], bfr[nt], acc[mt][nt]);
        }
    }
    __syncthreads();                           // all waves done with tile buffers

    // Stage f32 tile to LDS so RoPE can pair adjacent columns.
#pragma unroll
    for (int mt = 0; mt < 2; ++mt)
#pragma unroll
        for (int nt = 0; nt < 4; ++nt)
#pragma unroll
            for (int r = 0; r < 8; ++r) {
                int row = wm * 32 + mt * 16 + r + half8;
                int col = wn * 64 + nt * 16 + lane16;
                Cst[row * 136 + col] = acc[mt][nt][r];
            }
    __syncthreads();

#pragma unroll
    for (int e = 0; e < 32; ++e) {
        int p = e * 256 + t;                 // 8192 pairs
        int row = p >> 6, pc = p & 63;
        int col = pc * 2;
        int gr = rowBase + row, gc = colBase + col;
        int s = gr & (Sq - 1);
        float a  = Cst[row * 136 + col];
        float b2 = Cst[row * 136 + col + 1];
        float o0 = a, o1 = b2;
        if (applyRope) {
            int fi = (gc & (HDm - 1)) >> 1;
            float c  = cosp[s * (HDm / 2) + fi];
            float sn = sinp[s * (HDm / 2) + fi];
            o0 = a * c - b2 * sn;
            o1 = a * sn + b2 * c;
        }
        int head = gc >> 7;
        int bidx = gr >> 10;
        size_t o = ((size_t)(bidx * nheads + head) * Sq + s) * HDm + (gc & (HDm - 1));
        out[o]     = f2bf(o0);
        out[o + 1] = f2bf(o1);
    }
}

// ---------------------------------------------------------------------------
// Kernel 2: causal flash attention, double-buffered K/V staging.
// One block = (b, h, 128 Q rows); 8 waves x 16 Q rows, 16x128 f32 accum.
// ---------------------------------------------------------------------------
__device__ __forceinline__ void attn_issue(const bf16_t* __restrict__ kbase,
                                           const bf16_t* __restrict__ vbase,
                                           bf16_t* Ksb, bf16_t* Vtb,
                                           int j, int t)
{
    // K tile: 32x128 bf16, contiguous -> async b128 copies
    const bf16_t* ksrc = kbase + (size_t)j * HDm;
    int off = t * 16;
    async_cp16(ksrc + off,     Ksb + off);
    async_cp16(ksrc + off + 8, Ksb + off + 8);
    // V tile transposed into Vt[hd][kv]: vector row loads + b16 LDS scatter
    const bf16_t* vsrc = vbase + (size_t)j * HDm;
    int kv  = t >> 3;
    int hd0 = (t & 7) * 16;
    v8bf a = *(const v8bf*)(vsrc + kv * HDm + hd0);
    v8bf b = *(const v8bf*)(vsrc + kv * HDm + hd0 + 8);
#pragma unroll
    for (int i = 0; i < 8; ++i) {
        Vtb[(hd0 + i) * 32 + kv]     = a[i];
        Vtb[(hd0 + 8 + i) * 32 + kv] = b[i];
    }
}

__global__ __launch_bounds__(256) void flash_attn(
    const bf16_t* __restrict__ Q, const bf16_t* __restrict__ K,
    const bf16_t* __restrict__ V, bf16_t* __restrict__ O)
{
    __shared__ __align__(16) bf16_t Ks[2][32 * 128];   // [kv][hd]
    __shared__ __align__(16) bf16_t Vt[2][128 * 32];   // [hd][kv]
    __shared__ __align__(16) bf16_t Ps[8][16 * 32];    // per-wave P staging

    const int t = threadIdx.x;
    const int w = t >> 5, l = t & 31;
    const int lane16 = l & 15;
    const int half8  = ((l >> 4) & 1) * 8;
    const int half16 = ((l >> 4) & 1) * 16;
    const int qb = blockIdx.x * 128;
    const int h  = blockIdx.y;
    const int b  = blockIdx.z;
    const int kvh = h / REP;

    const bf16_t* qrowp = Q + ((size_t)(b * NH + h) * Sq + qb + w * 16 + lane16) * HDm;
    v16bf qa[4];
#pragma unroll
    for (int c = 0; c < 4; ++c) qa[c] = ldA(qrowp + c * 32, half8);

    const v8f vzero = {0.f,0.f,0.f,0.f,0.f,0.f,0.f,0.f};
    v8f o[8];
#pragma unroll
    for (int tt = 0; tt < 8; ++tt) o[tt] = vzero;
    float m_r[8], l_r[8];
#pragma unroll
    for (int r = 0; r < 8; ++r) { m_r[r] = -3.0e38f; l_r[r] = 0.f; }

    const bf16_t* kbase = K + (size_t)(b * NKV + kvh) * Sq * HDm;
    const bf16_t* vbase = V + (size_t)(b * NKV + kvh) * Sq * HDm;
    const float scale = 0.08838834764831845f;   // 1/sqrt(128)

    attn_issue(kbase, vbase, Ks[0], Vt[0], 0, t);
    const int NIT = (qb + 128) >> 5;
    for (int it = 0; it < NIT; ++it) {
        int j = it << 5, cur = it & 1;
        async_wait_all();
        __syncthreads();
        if (it + 1 < NIT) attn_issue(kbase, vbase, Ks[cur ^ 1], Vt[cur ^ 1], j + 32, t);

        // S = Q K^T  (two 16x16 logit tiles, K-chained over HD)
        v8f s0 = vzero, s1 = vzero;
#pragma unroll
        for (int c = 0; c < 4; ++c) {
            v16bf b0 = ldB(&Ks[cur][lane16 * 128 + c * 32], half16);
            s0 = wmma_bf16(qa[c], b0, s0);
            v16bf b1 = ldB(&Ks[cur][(16 + lane16) * 128 + c * 32], half16);
            s1 = wmma_bf16(qa[c], b1, s1);
        }

        // online softmax update, one row per (r, half)
#pragma unroll
        for (int r = 0; r < 8; ++r) {
            int qrow = qb + w * 16 + r + half8;
            float v0 = s0[r] * scale, v1 = s1[r] * scale;
            if (j + lane16 > qrow)       v0 = -3.0e38f;
            if (j + 16 + lane16 > qrow)  v1 = -3.0e38f;
            float mx = fmaxf(v0, v1);
#pragma unroll
            for (int mk = 1; mk < 16; mk <<= 1)
                mx = fmaxf(mx, __shfl_xor(mx, mk, 32));
            float mnew = fmaxf(m_r[r], mx);
            v0 = __expf(v0 - mnew);
            v1 = __expf(v1 - mnew);
            float rs = v0 + v1;
#pragma unroll
            for (int mk = 1; mk < 16; mk <<= 1)
                rs += __shfl_xor(rs, mk, 32);
            float alpha = __expf(m_r[r] - mnew);
            l_r[r] = l_r[r] * alpha + rs;
            m_r[r] = mnew;
#pragma unroll
            for (int tt = 0; tt < 8; ++tt) o[tt][r] *= alpha;
            Ps[w][(r + half8) * 32 + lane16]      = f2bf(v0);
            Ps[w][(r + half8) * 32 + 16 + lane16] = f2bf(v1);
        }

        // O += P V   (DS in-order within wave: staging readable w/o barrier)
        v16bf pa = ldA(&Ps[w][lane16 * 32], half8);
#pragma unroll
        for (int tt = 0; tt < 8; ++tt) {
            v16bf vb = ldB(&Vt[cur][(tt * 16 + lane16) * 32], half16);
            o[tt] = wmma_bf16(pa, vb, o[tt]);
        }
    }

    // epilogue: O / l, bf16, layout (B*S, H*HD)
#pragma unroll
    for (int r = 0; r < 8; ++r) {
        float inv = 1.0f / l_r[r];
        int qrow = qb + w * 16 + r + half8;
        size_t rb = ((size_t)b * Sq + qrow) * (NH * HDm) + (size_t)h * HDm;
#pragma unroll
        for (int tt = 0; tt < 8; ++tt)
            O[rb + tt * 16 + lane16] = f2bf(o[tt][r] * inv);
    }
}

// ---------------------------------------------------------------------------
// Kernel 3: out = attn(bf16) @ Wo_t(bf16,[N][K]) -> fp32
// ---------------------------------------------------------------------------
__global__ __launch_bounds__(256) void out_gemm(
    const bf16_t* __restrict__ A, const bf16_t* __restrict__ Wt,
    float* __restrict__ Out)
{
    __shared__ __align__(16) unsigned char smem[65536];

    const int t = threadIdx.x;
    const int w = t >> 5, l = t & 31;
    const int wm = w & 3, wn = w >> 2;
    const int lane16 = l & 15;
    const int half8  = ((l >> 4) & 1) * 8;
    const int half16 = ((l >> 4) & 1) * 16;
    const int rowBase = blockIdx.x * 128;
    const int colBase = blockIdx.y * 128;
    const bf16_t* Ab = A  + (size_t)rowBase * Dm;
    const bf16_t* Bb = Wt + (size_t)colBase * Dm;

    v8f acc[2][4];
    const v8f vzero = {0.f,0.f,0.f,0.f,0.f,0.f,0.f,0.f};
#pragma unroll
    for (int mt = 0; mt < 2; ++mt)
#pragma unroll
        for (int nt = 0; nt < 4; ++nt) acc[mt][nt] = vzero;

    gemm_issue(Ab, Bb, smem, 0, t, 0);
    const int NIT = Dm / 64;
    for (int it = 0; it < NIT; ++it) {
        int cur = it & 1;
        async_wait_all();
        __syncthreads();
        if (it + 1 < NIT) gemm_issue(Ab, Bb, smem, cur ^ 1, t, (it + 1) * 64);

        const bf16_t* Asb = (const bf16_t*)(smem + cur * 32768);
        const bf16_t* Bsb = (const bf16_t*)(smem + cur * 32768 + 16384);
#pragma unroll
        for (int kk = 0; kk < 2; ++kk) {
            v16bf af[2], bfr[4];
#pragma unroll
            for (int mt = 0; mt < 2; ++mt)
                af[mt] = ldA(&Asb[(wm * 32 + mt * 16 + lane16) * 64 + kk * 32], half8);
#pragma unroll
            for (int nt = 0; nt < 4; ++nt)
                bfr[nt] = ldB(&Bsb[(wn * 64 + nt * 16 + lane16) * 64 + kk * 32], half16);
#pragma unroll
            for (int mt = 0; mt < 2; ++mt)
#pragma unroll
                for (int nt = 0; nt < 4; ++nt)
                    acc[mt][nt] = wmma_bf16(af[mt], bfr[nt], acc[mt][nt]);
        }
    }

#pragma unroll
    for (int mt = 0; mt < 2; ++mt)
#pragma unroll
        for (int nt = 0; nt < 4; ++nt)
#pragma unroll
            for (int r = 0; r < 8; ++r) {
                int gr = rowBase + wm * 32 + mt * 16 + r + half8;
                int gc = colBase + wn * 64 + nt * 16 + lane16;
                Out[(size_t)gr * Dm + gc] = acc[mt][nt][r];
            }
}

// ---------------------------------------------------------------------------
extern "C" void kernel_launch(void* const* d_in, const int* in_sizes, int n_in,
                              void* d_out, int out_size, void* d_ws, size_t ws_size,
                              hipStream_t stream) {
    (void)in_sizes; (void)n_in; (void)out_size; (void)ws_size;
    const float* seq  = (const float*)d_in[0];
    const float* Wq   = (const float*)d_in[1];
    const float* Wk   = (const float*)d_in[2];
    const float* Wv   = (const float*)d_in[3];
    const float* Wo   = (const float*)d_in[4];
    const float* cosp = (const float*)d_in[5];
    const float* sinp = (const float*)d_in[6];
    // d_in[7] = mask (causal, applied analytically), d_in[8] = start_position (0)

    const size_t nBig = (size_t)Dm * Dm;        // 16.7M
    const size_t nKV  = (size_t)Dm * (NKV*HDm); // 4.2M
    bf16_t* seq_b = (bf16_t*)d_ws;
    bf16_t* wq_t  = seq_b + nBig;
    bf16_t* wk_t  = wq_t  + nBig;
    bf16_t* wv_t  = wk_t  + nKV;
    bf16_t* wo_t  = wv_t  + nKV;
    bf16_t* q_ws  = wo_t  + nBig;
    bf16_t* k_ws  = q_ws  + (size_t)Bz * NH  * Sq * HDm;
    bf16_t* v_ws  = k_ws  + (size_t)Bz * NKV * Sq * HDm;
    bf16_t* a_ws  = v_ws  + (size_t)Bz * NKV * Sq * HDm;
    float* out = (float*)d_out;

    dim3 blk(256);
    cvt_bf16<<<dim3(nBig / 2048), blk, 0, stream>>>(seq, seq_b);
    cvt_transpose<<<dim3(64, 64), blk, 0, stream>>>(Wq, wq_t, Dm, Dm);
    cvt_transpose<<<dim3(64, 16), blk, 0, stream>>>(Wk, wk_t, Dm, NKV * HDm);
    cvt_transpose<<<dim3(64, 16), blk, 0, stream>>>(Wv, wv_t, Dm, NKV * HDm);
    cvt_transpose<<<dim3(64, 64), blk, 0, stream>>>(Wo, wo_t, Dm, Dm);

    qkv_gemm_rope<<<dim3(32, 32), blk, 0, stream>>>(seq_b, wq_t, cosp, sinp, q_ws, NH,  1);
    qkv_gemm_rope<<<dim3(32,  8), blk, 0, stream>>>(seq_b, wk_t, cosp, sinp, k_ws, NKV, 1);
    qkv_gemm_rope<<<dim3(32,  8), blk, 0, stream>>>(seq_b, wv_t, cosp, sinp, v_ws, NKV, 0);
    flash_attn<<<dim3(Sq / 128, NH, Bz), blk, 0, stream>>>(q_ws, k_ws, v_ws, a_ws);
    out_gemm<<<dim3(32, 32), blk, 0, stream>>>(a_ws, wo_t, out);
}
